// TimeAttnBlock_91087666413726
// MI455X (gfx1250) — compile-verified, matrix operand order checked by reading, not run on hardware
//
#include <hip/hip_runtime.h>
#include <hip/hip_bf16.h>

// ---------------------------------------------------------------------------
// TimeAttnBlock for MI455X (gfx1250): GroupNorm -> q/k/v proj -> softmax attn
// -> o proj + residual.  All GEMMs via v_wmma_f32_16x16x32_bf16 (wave32).
// Tile staging uses GLOBAL_LOAD_ASYNC_TO_LDS_B128 (ASYNCcnt) with double
// buffering so DMA overlaps WMMA compute.
// B=8, C=512, T=2048, 32 groups.
// ---------------------------------------------------------------------------

#define CDIM 512
#define TDIM 2048
#define BDIM 8
#define NGRP 32
#define CPG  16   // channels per group

typedef __attribute__((ext_vector_type(16))) __bf16 v16bf;
typedef __attribute__((ext_vector_type(8)))  float  v8f;

// Load a 16-element bf16 fragment (32 bytes) as two 16B vectors.
static __device__ __forceinline__ v16bf ld_frag(const __bf16* p) {
    union { uint4 u[2]; v16bf v; } cv;
    cv.u[0] = *reinterpret_cast<const uint4*>(p);
    cv.u[1] = *reinterpret_cast<const uint4*>(p + 8);
    return cv.v;
}

static __device__ __forceinline__ v8f wmma_bf16(v16bf a, v16bf b, v8f c) {
    // D = A(16x32) * B(32x16) + C(16x16), f32 accumulate
    return __builtin_amdgcn_wmma_f32_16x16x32_bf16(
        /*neg_a=*/false, a, /*neg_b=*/false, b,
        /*c_mod=*/(short)0, c, /*reuse_a=*/false, /*reuse_b=*/false);
}

// Low 32 bits of a generic pointer into LDS == LDS byte address (ISA 10.2).
static __device__ __forceinline__ unsigned lds_addr(const void* p) {
    return (unsigned)(unsigned long long)(p);
}

// Async copy of 32 bytes (16 bf16) global -> LDS, per lane.  INST_OFFSET is
// added to BOTH the LDS and global addresses, so one pair covers both halves.
// Tracked by ASYNCcnt (2 instructions per call).
static __device__ __forceinline__ void async_cp32(unsigned loff, const __bf16* g) {
    asm volatile("global_load_async_to_lds_b128 %0, %1, off\n\t"
                 "global_load_async_to_lds_b128 %0, %1, off offset:16"
                 :: "v"(loff), "v"(g) : "memory");
}

static __device__ __forceinline__ void wait_async0() {
    asm volatile("s_wait_asynccnt 0x0" ::: "memory");
}
static __device__ __forceinline__ void wait_async4() {
    asm volatile("s_wait_asynccnt 0x4" ::: "memory");
}

// ---------------------------------------------------------------------------
// Kernel 0: convert the four 512x512 f32 weight matrices to bf16.
// ---------------------------------------------------------------------------
__global__ void cvt_weights(const float* __restrict__ qw, const float* __restrict__ kw,
                            const float* __restrict__ vw, const float* __restrict__ ow,
                            __bf16* __restrict__ dst) {
    const int n = CDIM * CDIM;
    int i = blockIdx.x * blockDim.x + threadIdx.x;
    if (i < n) {
        dst[i]         = (__bf16)qw[i];
        dst[n + i]     = (__bf16)kw[i];
        dst[2 * n + i] = (__bf16)vw[i];
        dst[3 * n + i] = (__bf16)ow[i];
    }
}

// ---------------------------------------------------------------------------
// Kernel 1: GroupNorm.  One block per (batch, group) = 16 channels x 2048 T.
// Writes hT in [B][T][C] bf16 (t-major: makes WMMA B-fragments contiguous).
// ---------------------------------------------------------------------------
__global__ void groupnorm_kernel(const float* __restrict__ x,
                                 const float* __restrict__ gamma,
                                 const float* __restrict__ beta,
                                 __bf16* __restrict__ hT) {
    __shared__ float s1[256];
    __shared__ float s2[256];
    const int b = blockIdx.x >> 5;
    const int g = blockIdx.x & 31;
    const size_t base = (size_t)b * CDIM * TDIM + (size_t)g * CPG * TDIM;
    const int N = CPG * TDIM;  // 32768
    float sum = 0.f, sumsq = 0.f;
    for (int i = threadIdx.x; i < N; i += 256) {
        float v = x[base + i];
        sum += v;
        sumsq += v * v;
    }
    s1[threadIdx.x] = sum;
    s2[threadIdx.x] = sumsq;
    __syncthreads();
    for (int off = 128; off > 0; off >>= 1) {
        if (threadIdx.x < (unsigned)off) {
            s1[threadIdx.x] += s1[threadIdx.x + off];
            s2[threadIdx.x] += s2[threadIdx.x + off];
        }
        __syncthreads();
    }
    const float mean = s1[0] * (1.0f / N);
    const float var  = s2[0] * (1.0f / N) - mean * mean;
    const float inv  = rsqrtf(var + 1e-6f);
    for (int i = threadIdx.x; i < N; i += 256) {
        const int cl = i >> 11;          // local channel
        const int t  = i & (TDIM - 1);
        const int c  = g * CPG + cl;
        const float v = (x[base + i] - mean) * inv * gamma[c] + beta[c];
        hT[(size_t)b * TDIM * CDIM + (size_t)t * CDIM + c] = (__bf16)v;
    }
}

// ---------------------------------------------------------------------------
// Kernel 2: bf16 WMMA GEMM.  out[m,n] = sum_k W[m,k] * Bt[n,k] + bias[m]
//   W  : [512][512] bf16 row-major (M,K)
//   Bt : per-batch [2048][512] bf16 (N-major, K contiguous)
// mode 0: bf16 out [M][N]      (v: ready A-matrix for P*V)
// mode 1: bf16 out [N][M]      (qT/kT: ready A/B matrices for scores)
// mode 2: f32  out [M][N] + residual (final o-projection into d_out)
// Block tile 128x128, K-step 32, 8 waves (2x4), each wave 4x2 WMMA tiles.
// Staging: double-buffered GLOBAL_LOAD_ASYNC_TO_LDS_B128.
// ---------------------------------------------------------------------------
__global__ void gemm_wmma_kernel(const __bf16* __restrict__ W,
                                 const __bf16* __restrict__ Bt,
                                 const float*  __restrict__ bias,
                                 const float*  __restrict__ resid,
                                 __bf16* __restrict__ outB,
                                 float*  __restrict__ outF,
                                 int mode) {
    __shared__ __bf16 sA[2][128 * 32];
    __shared__ __bf16 sB[2][128 * 32];
    const int b  = blockIdx.z;
    const int n0 = blockIdx.x * 128;
    const int m0 = blockIdx.y * 128;
    const size_t bb = (size_t)b * TDIM * CDIM;
    const int tid  = threadIdx.x;
    const int lane = tid & 31;
    const int wid  = tid >> 5;
    const int wm = (wid & 1) * 64;     // wave M offset in block tile
    const int wn = (wid >> 1) * 32;    // wave N offset
    const int ml = lane & 15;
    const int kb = (lane >> 4) * 16;   // K-half select: contiguous k = 16h + s, same for A and B
    const int mh = (lane >> 4) * 8;    // C/D M-half select

    v8f acc[4][2];
#pragma unroll
    for (int i = 0; i < 4; i++)
#pragma unroll
        for (int j = 0; j < 2; j++) acc[i][j] = (v8f){};

    const int ldr = tid >> 1;          // staging row 0..127
    const int ldk = (tid & 1) * 16;    // staging K half
    const __bf16* gA = &W[(size_t)(m0 + ldr) * CDIM + ldk];
    const __bf16* gB = &Bt[bb + (size_t)(n0 + ldr) * CDIM + ldk];
    const unsigned lA0 = lds_addr(&sA[0][ldr * 32 + ldk]);
    const unsigned lB0 = lds_addr(&sB[0][ldr * 32 + ldk]);
    const unsigned lA1 = lds_addr(&sA[1][ldr * 32 + ldk]);
    const unsigned lB1 = lds_addr(&sB[1][ldr * 32 + ldk]);

    // Prologue: stage K-tile 0 into buffer 0 (4 async b128 per thread).
    async_cp32(lA0, gA);
    async_cp32(lB0, gB);

    int buf = 0;
    for (int kc = 0; kc < CDIM; kc += 32) {
        if (kc + 32 < CDIM) {
            // Kick next tile's DMA into the other buffer, then wait for the
            // current tile only (async loads complete in order).
            async_cp32(buf ? lA0 : lA1, gA + kc + 32);
            async_cp32(buf ? lB0 : lB1, gB + kc + 32);
            wait_async4();
        } else {
            wait_async0();
        }
        __syncthreads();

        const __bf16* tA = sA[buf];
        const __bf16* tB = sB[buf];
        v16bf bfr[2];
#pragma unroll
        for (int j = 0; j < 2; j++)
            bfr[j] = ld_frag(&tB[(wn + j * 16 + ml) * 32 + kb]);
#pragma unroll
        for (int i = 0; i < 4; i++) {
            v16bf af = ld_frag(&tA[(wm + i * 16 + ml) * 32 + kb]);
#pragma unroll
            for (int j = 0; j < 2; j++) acc[i][j] = wmma_bf16(af, bfr[j], acc[i][j]);
        }
        __syncthreads();
        buf ^= 1;
    }

    // Epilogue: C/D layout is lane[0:15]->M=r, lane[16:31]->M=8+r, N=lane&15.
#pragma unroll
    for (int i = 0; i < 4; i++) {
#pragma unroll
        for (int j = 0; j < 2; j++) {
#pragma unroll
            for (int r = 0; r < 8; r++) {
                const int m = m0 + wm + i * 16 + mh + r;
                const int n = n0 + wn + j * 16 + ml;
                const float v = acc[i][j][r] + bias[m];
                if (mode == 0) {
                    outB[bb + (size_t)m * TDIM + n] = (__bf16)v;
                } else if (mode == 1) {
                    outB[bb + (size_t)n * CDIM + m] = (__bf16)v;
                } else {
                    const size_t o = bb + (size_t)m * TDIM + n;
                    outF[o] = v + resid[o];
                }
            }
        }
    }
}

// ---------------------------------------------------------------------------
// Kernel 3: fused attention.  One workgroup per (batch, 16-query tile).
// Phase 1: 8 waves compute disjoint 16-wide s-tiles of S = (qT * kT^T)/sqrt(C)
//          via WMMA (K=512), store exp-ready scores in LDS (bf16 [16][2048]).
// Phase 2: block softmax (rowmax, exp, rowsum) in place.
// Phase 3: each wave owns 64 output channels; accumulates V * P^T with WMMA
//          (P rows in LDS are directly valid B-fragments), scales by 1/rowsum.
// Dynamic LDS: 16*2048*2 + 16*512*2 + (256+32)*4 = 83072 bytes.
// ---------------------------------------------------------------------------
__global__ void attn_wmma_kernel(const __bf16* __restrict__ qT,
                                 const __bf16* __restrict__ kT,
                                 const __bf16* __restrict__ vM,
                                 __bf16* __restrict__ attnT) {
    extern __shared__ char smem[];
    __bf16* sS   = (__bf16*)smem;                                   // [16][2048]
    __bf16* sQ   = (__bf16*)(smem + (size_t)16 * TDIM * 2);         // [16][512]
    float*  sRed = (float*)(smem + (size_t)16 * TDIM * 2 + (size_t)16 * CDIM * 2);
    float*  sMax = sRed + 256;                                      // [16]
    float*  sSum = sMax + 16;                                       // [16]

    const int b  = blockIdx.y;
    const int t0 = blockIdx.x * 16;
    const size_t bb = (size_t)b * TDIM * CDIM;
    const int tid  = threadIdx.x;
    const int lane = tid & 31;
    const int wid  = tid >> 5;
    const int ml = lane & 15;
    const int kb = (lane >> 4) * 16;
    const int mh = (lane >> 4) * 8;
    const float scale = 0.044194173824159216f;  // 512^-0.5

    // Stage the 16x512 query tile (qT is [T][C]) via async copies.
    for (int i = tid * 16; i < 16 * CDIM; i += 256 * 16)
        async_cp32(lds_addr(&sQ[i]), &qT[bb + (size_t)t0 * CDIM + i]);
    wait_async0();
    __syncthreads();

    // Phase 1: scores.  A = q (M=t, K=c) from LDS; B = k (N=s, K=c) from global.
    for (int st = wid; st < TDIM / 16; st += 8) {
        const int s0 = st * 16;
        v8f acc = (v8f){};
#pragma unroll 4
        for (int kc = 0; kc < CDIM; kc += 32) {
            v16bf af = ld_frag(&sQ[ml * CDIM + kc + kb]);
            const __bf16* gk = &kT[bb + (size_t)(s0 + ml) * CDIM + kc + kb];
            if (kc + 32 < CDIM) __builtin_prefetch(gk + 32, 0, 1);
            v16bf bf = ld_frag(gk);
            acc = wmma_bf16(af, bf, acc);
        }
#pragma unroll
        for (int r = 0; r < 8; r++)
            sS[(mh + r) * TDIM + s0 + ml] = (__bf16)(acc[r] * scale);
    }
    __syncthreads();

    // Phase 2: softmax over s for each of the 16 query rows (16 threads/row).
    {
        const int row = tid >> 4;
        const int sub = tid & 15;
        float mx = -3.0e38f;
        for (int s = sub; s < TDIM; s += 16)
            mx = fmaxf(mx, (float)sS[row * TDIM + s]);
        sRed[tid] = mx;
        __syncthreads();
        if (tid < 16) {
            float m2 = sRed[tid * 16];
            for (int j = 1; j < 16; j++) m2 = fmaxf(m2, sRed[tid * 16 + j]);
            sMax[tid] = m2;
        }
        __syncthreads();
        const float rm = sMax[row];
        float sum = 0.f;
        for (int s = sub; s < TDIM; s += 16) {
            const float e = __expf((float)sS[row * TDIM + s] - rm);
            sS[row * TDIM + s] = (__bf16)e;
            sum += e;
        }
        sRed[tid] = sum;
        __syncthreads();
        if (tid < 16) {
            float t2 = 0.f;
            for (int j = 0; j < 16; j++) t2 += sRed[tid * 16 + j];
            sSum[tid] = t2;
        }
        __syncthreads();
    }

    // Phase 3: out[c, t0..t0+15] = V * P^T.  Wave owns channels [wid*64, +64).
    const int c0 = wid * 64;
    v8f oacc[4];
#pragma unroll
    for (int i = 0; i < 4; i++) oacc[i] = (v8f){};
    for (int s0 = 0; s0 < TDIM; s0 += 32) {
        // B fragment: N = t (= ml), K = s -> contiguous row of sS.
        v16bf pf = ld_frag(&sS[ml * TDIM + s0 + kb]);
#pragma unroll
        for (int i = 0; i < 4; i++) {
            // A fragment: M = c (= ml), K = s -> contiguous row of vM [C][T].
            v16bf af = ld_frag(&vM[bb + (size_t)(c0 + i * 16 + ml) * TDIM + s0 + kb]);
            oacc[i] = wmma_bf16(af, pf, oacc[i]);
        }
    }
    const float inv = 1.0f / sSum[ml];
#pragma unroll
    for (int i = 0; i < 4; i++)
#pragma unroll
        for (int r = 0; r < 8; r++)
            attnT[bb + (size_t)(t0 + ml) * CDIM + c0 + i * 16 + mh + r] =
                (__bf16)(oacc[i][r] * inv);
}

// ---------------------------------------------------------------------------
// Host launch
// ---------------------------------------------------------------------------
extern "C" void kernel_launch(void* const* d_in, const int* in_sizes, int n_in,
                              void* d_out, int out_size, void* d_ws, size_t ws_size,
                              hipStream_t stream) {
    (void)in_sizes; (void)n_in; (void)out_size; (void)ws_size;
    const float* x     = (const float*)d_in[0];
    const float* gamma = (const float*)d_in[1];
    const float* beta  = (const float*)d_in[2];
    const float* q_w   = (const float*)d_in[3];
    const float* q_b   = (const float*)d_in[4];
    const float* k_w   = (const float*)d_in[5];
    const float* k_b   = (const float*)d_in[6];
    const float* v_w   = (const float*)d_in[7];
    const float* v_b   = (const float*)d_in[8];
    const float* o_w   = (const float*)d_in[9];
    const float* o_b   = (const float*)d_in[10];
    float* out = (float*)d_out;

    const size_t WSZ = (size_t)CDIM * CDIM;        // 262144 elems per weight
    const size_t ASZ = (size_t)BDIM * TDIM * CDIM; // 8388608 elems per activation
    __bf16* w_bf  = (__bf16*)d_ws;       // [4][C][C]
    __bf16* hT    = w_bf + 4 * WSZ;      // [B][T][C]
    __bf16* qTb   = hT + ASZ;            // [B][T][C]
    __bf16* kTb   = qTb + ASZ;           // [B][T][C]
    __bf16* vMb   = kTb + ASZ;           // [B][C][T]
    __bf16* attnT = vMb + ASZ;           // [B][T][C]   (~86 MB total)

    cvt_weights<<<dim3((unsigned)((WSZ + 255) / 256)), 256, 0, stream>>>(
        q_w, k_w, v_w, o_w, w_bf);
    groupnorm_kernel<<<dim3(BDIM * NGRP), 256, 0, stream>>>(x, gamma, beta, hT);

    dim3 gg(TDIM / 128, CDIM / 128, BDIM);
    gemm_wmma_kernel<<<gg, 256, 0, stream>>>(w_bf,           hT,    q_b, nullptr, qTb,   nullptr, 1);
    gemm_wmma_kernel<<<gg, 256, 0, stream>>>(w_bf + WSZ,     hT,    k_b, nullptr, kTb,   nullptr, 1);
    gemm_wmma_kernel<<<gg, 256, 0, stream>>>(w_bf + 2 * WSZ, hT,    v_b, nullptr, vMb,   nullptr, 0);

    const size_t smem = (size_t)16 * TDIM * 2 + (size_t)16 * CDIM * 2 + (256 + 32) * 4;
    attn_wmma_kernel<<<dim3(TDIM / 16, BDIM), 256, smem, stream>>>(qTb, kTb, vMb, attnT);

    gemm_wmma_kernel<<<gg, 256, 0, stream>>>(w_bf + 3 * WSZ, attnT, o_b, x,       nullptr, out,    2);
}
